// Ernie4_5_VLAttention_25933012533746
// MI455X (gfx1250) — compile-verified, matrix operand order checked by reading, not run on hardware
//
#include <hip/hip_runtime.h>
#include <math.h>

#define T_SEQ     2048
#define HIDDEN_D  2048
#define NH        16
#define NKV       4
#define HD        128
#define QKV_COLS  3072   // (16 + 2*4) * 128

typedef __attribute__((ext_vector_type(16))) __bf16 v16bf;
typedef __attribute__((ext_vector_type(8)))  float  v8f;

union FragU { v16bf v; unsigned u[8]; };

// ---- CDNA5 async copy: global -> LDS, 16B per lane, tracked by ASYNCcnt ----
__device__ __forceinline__ void async_b128(unsigned lds_byte_off,
                                           const void* sbase, unsigned gbyte_off) {
  asm volatile("global_load_async_to_lds_b128 %0, %1, %2"
               :: "v"(lds_byte_off), "v"(gbyte_off), "s"(sbase) : "memory");
}
#define WAIT_ASYNCCNT_0  asm volatile("s_wait_asynccnt 0x0" ::: "memory")
#define WAIT_ASYNCCNT_8  asm volatile("s_wait_asynccnt 0x8" ::: "memory")

__device__ __forceinline__ unsigned lds_off_u32(const void* p) {
  return (unsigned)(size_t)p;   // low 32 bits of flat shared addr = LDS offset
}

// Load a 16x32 bf16 fragment (A layout; also B^T layout with row=n) from LDS.
// ISA 16-bit A map: lane m=lane&15, half kh=lane>>4; VGPR j holds
// K = (j/4)*16 + kh*8 + (j%4)*2 (+1). `ld` must be even.
__device__ __forceinline__ v16bf load_frag_bf16(const __bf16* tile, int ld,
                                                int row0, int col0, int lane) {
  const int m  = lane & 15;
  const int kh = (lane >> 4) & 1;
  const __bf16* p = tile + (row0 + m) * ld + col0;
  FragU f;
#pragma unroll
  for (int j = 0; j < 8; ++j) {
    const int k = ((j >> 2) << 4) + (kh << 3) + ((j & 3) << 1);
    f.u[j] = *(const unsigned*)(p + k);
  }
  return f.v;
}

// fp32-accuracy product via bf16 hi/lo split: (ah+al)(bh+bl) ~= ah*bh+ah*bl+al*bh
__device__ __forceinline__ v8f wmma_split(v16bf ah, v16bf al, v16bf bh, v16bf bl, v8f c) {
  c = __builtin_amdgcn_wmma_f32_16x16x32_bf16(false, ah, false, bh, (short)0, c, false, false);
  c = __builtin_amdgcn_wmma_f32_16x16x32_bf16(false, ah, false, bl, (short)0, c, false, false);
  c = __builtin_amdgcn_wmma_f32_16x16x32_bf16(false, al, false, bh, (short)0, c, false, false);
  return c;
}

__device__ __forceinline__ void split_bf16(float x, __bf16& h, __bf16& l) {
  h = (__bf16)x;
  l = (__bf16)(x - (float)h);
}

// ---------------- operand preparation kernels (one-pass, memory-bound) -------
__global__ __launch_bounds__(256) void split_kernel(const float* __restrict__ src,
                                                    __bf16* __restrict__ hi,
                                                    __bf16* __restrict__ lo, int n) {
  int i = blockIdx.x * 256 + threadIdx.x;
  if (i < n) { __bf16 h, l; split_bf16(src[i], h, l); hi[i] = h; lo[i] = l; }
}

// dst[c][r] = split(src[r][c]); src is R x C (ld src_ld), dst is C x R (ld dst_ld)
__global__ __launch_bounds__(256) void transpose_split_kernel(
    const float* __restrict__ src, int src_ld,
    __bf16* __restrict__ hi, __bf16* __restrict__ lo, int dst_ld, int R, int C) {
  __shared__ float tile[32][33];
  const int tid = threadIdx.x;
  const int r0 = blockIdx.y * 32, c0 = blockIdx.x * 32;
  const int ci = tid & 31, rg = tid >> 5;
#pragma unroll
  for (int i = 0; i < 4; ++i) {
    int r = rg + i * 8;
    tile[r][ci] = src[(size_t)(r0 + r) * src_ld + c0 + ci];
  }
  __syncthreads();
#pragma unroll
  for (int i = 0; i < 4; ++i) {
    int cj = rg + i * 8, rr = ci;
    __bf16 h, l; split_bf16(tile[rr][cj], h, l);
    size_t idx = (size_t)(c0 + cj) * dst_ld + r0 + rr;
    hi[idx] = h; lo[idx] = l;
  }
}

// ---------------- async double-buffered bf16-split WMMA GEMM -----------------
// C(MxN,f32) = A(MxK) * Bt(NxK)^T ; A,Bt pre-split bf16 hi/lo, row stride K.
// Block: 256 thr = 8 waves (4M x 2N); block tile 128x128, BK=32; wave 32x64.
__global__ __launch_bounds__(256) void gemm_bf16_async_kernel(
    const __bf16* __restrict__ Ah, const __bf16* __restrict__ Al,
    const __bf16* __restrict__ Bth, const __bf16* __restrict__ Btl,
    float* __restrict__ C, int M, int N, int K, int ldc) {
  __shared__ __align__(16) __bf16 smem[2 * 4 * 128 * 40];  // 2 bufs x {Ah,Al,Bh,Bl}[128][40]
  const int tid  = threadIdx.x;
  const int lane = tid & 31;
  const int wave = tid >> 5;
  const int wm = wave & 3, wn = wave >> 2;
  const int m0 = blockIdx.y * 128, n0 = blockIdx.x * 128;
  const unsigned smbase = lds_off_u32(smem);
  const __bf16* gsrc[4] = { Ah, Al, Bth, Btl };

  auto issue = [&](int buf, int k0) {
    const unsigned base = smbase + (unsigned)buf * (4 * 128 * 40 * 2);
#pragma unroll
    for (int i = 0; i < 8; ++i) {
      const int c2 = i * 256 + tid;           // 0..2047 chunks of 16B
      const int tsel = c2 >> 9;               // 0:Ah 1:Al 2:Bh 3:Bl
      const int cc = c2 & 511;
      const int r = cc >> 2, q = cc & 3;      // row, 16B sub-chunk
      const int grow = ((tsel < 2) ? m0 : n0) + r;
      const unsigned goff = (unsigned)(((size_t)grow * K + k0 + q * 8) * 2);
      const unsigned loff = base + (unsigned)tsel * (128 * 40 * 2)
                                 + (unsigned)(r * 40 + q * 8) * 2;
      async_b128(loff, gsrc[tsel], goff);
    }
  };

  const int nk = K >> 5;
  issue(0, 0);
  v8f acc[2][4] = {};
  for (int kt = 0; kt < nk; ++kt) {
    if (kt + 1 < nk) { issue((kt + 1) & 1, (kt + 1) * 32); WAIT_ASYNCCNT_8; }
    else             { WAIT_ASYNCCNT_0; }
    __syncthreads();
    const __bf16* sb  = smem + (kt & 1) * (4 * 128 * 40);
    const __bf16* tAh = sb, *tAl = sb + 128 * 40;
    const __bf16* tBh = sb + 2 * 128 * 40, *tBl = sb + 3 * 128 * 40;
    v16bf afh[2], afl[2], bfh[4], bfl[4];
#pragma unroll
    for (int s = 0; s < 2; ++s) {
      afh[s] = load_frag_bf16(tAh, 40, wm * 32 + s * 16, 0, lane);
      afl[s] = load_frag_bf16(tAl, 40, wm * 32 + s * 16, 0, lane);
    }
#pragma unroll
    for (int t = 0; t < 4; ++t) {
      bfh[t] = load_frag_bf16(tBh, 40, wn * 64 + t * 16, 0, lane);
      bfl[t] = load_frag_bf16(tBl, 40, wn * 64 + t * 16, 0, lane);
    }
#pragma unroll
    for (int s = 0; s < 2; ++s)
#pragma unroll
      for (int t = 0; t < 4; ++t)
        acc[s][t] = wmma_split(afh[s], afl[s], bfh[t], bfl[t], acc[s][t]);
    __syncthreads();   // all waves done with buf before it is refilled
  }

  const int cn = lane & 15, ch = lane >> 4;
#pragma unroll
  for (int s = 0; s < 2; ++s)
#pragma unroll
    for (int t = 0; t < 4; ++t)
#pragma unroll
      for (int v = 0; v < 8; ++v) {
        int row = m0 + wm * 32 + s * 16 + v + ch * 8;
        int col = n0 + wn * 64 + t * 16 + cn;
        C[(size_t)row * ldc + col] = acc[s][t][v];
      }
}

// --------- mRoPE: rotate q/k in fp32, emit bf16 hi/lo Q and K directly -------
// Section rule: i<44 -> pos row (i odd ? 2 : 1); i>=44 -> row 0; freq idx = i.
__global__ __launch_bounds__(256) void rope_split_kernel(
    const int* __restrict__ positions, const float* __restrict__ qkv,
    __bf16* __restrict__ Qh, __bf16* __restrict__ Ql,
    __bf16* __restrict__ Kh, __bf16* __restrict__ Kl) {
  __shared__ float cs[64], sn[64];
  const int t = blockIdx.x, tid = threadIdx.x;
  if (tid < 64) {
    const int i   = tid;
    const int row = (i < 44) ? ((i & 1) ? 2 : 1) : 0;
    const float inv = powf(500000.0f, -(float)i / 64.0f);
    const float ang = (float)positions[row * T_SEQ + t] * inv;
    cs[i] = cosf(ang); sn[i] = sinf(ang);
  }
  __syncthreads();
  const float* rowp = qkv + (size_t)t * QKV_COLS;
  for (int p = tid; p < 1024; p += 256) {           // q: 16 heads * 64 pairs
    const int col = p * 2, i = (col & 127) >> 1;
    const float x1 = rowp[col], x2 = rowp[col + 1];
    __bf16 h, l;
    split_bf16(x1 * cs[i] - x2 * sn[i], h, l);
    Qh[(size_t)t * 2048 + col] = h;     Ql[(size_t)t * 2048 + col] = l;
    split_bf16(x2 * cs[i] + x1 * sn[i], h, l);
    Qh[(size_t)t * 2048 + col + 1] = h; Ql[(size_t)t * 2048 + col + 1] = l;
  }
  {
    const int p = tid;                               // k: 4 heads * 64 pairs
    const int col = p * 2, i = (col & 127) >> 1;
    const float x1 = rowp[2048 + col], x2 = rowp[2048 + col + 1];
    __bf16 h, l;
    split_bf16(x1 * cs[i] - x2 * sn[i], h, l);
    Kh[(size_t)t * 512 + col] = h;     Kl[(size_t)t * 512 + col] = l;
    split_bf16(x2 * cs[i] + x1 * sn[i], h, l);
    Kh[(size_t)t * 512 + col + 1] = h; Kl[(size_t)t * 512 + col + 1] = l;
  }
}

// ---------------- flash attention, async-staged bf16 operands ----------------
// Q:[T][2048] K:[T][512] Vt:[NKV*128][T] (hi/lo). Out: bf16 hi/lo [T][2048].
__global__ __launch_bounds__(256) void attn_kernel(
    const __bf16* __restrict__ Qh, const __bf16* __restrict__ Ql,
    const __bf16* __restrict__ Kh, const __bf16* __restrict__ Kl,
    const __bf16* __restrict__ Vth, const __bf16* __restrict__ Vtl,
    __bf16* __restrict__ Oh, __bf16* __restrict__ Ol) {
  __shared__ __align__(16) char smem[69632];
  __bf16* const sb   = (__bf16*)smem;
  __bf16* const sQh  = sb;                 // [128][136] (prologue only)
  __bf16* const sQl  = sb + 128 * 136;
  __bf16* const sKh  = sb;                 // [32][136]
  __bf16* const sKl  = sb + 4352;
  __bf16* const sVth = sb + 8704;          // [128][40] ([d][k])
  __bf16* const sVtl = sb + 13824;
  __bf16* const sP   = sb + 18944;         // 8 waves x {Ph,Pl}[16][34]

  const int tid  = threadIdx.x;
  const int lane = tid & 31;
  const int wave = tid >> 5;
  const int head = blockIdx.y;
  const int kvh  = head >> 2;
  const int q0   = blockIdx.x * 128;
  const int qr0  = q0 + wave * 16;
  const float scale = 0.088388347648318447f;  // 1/sqrt(128)
  const unsigned smbase = lds_off_u32(smem);

  // stage Q tile 128x128 hi/lo via async (16 chunks/thread)
#pragma unroll
  for (int i = 0; i < 16; ++i) {
    const int c2 = i * 256 + tid;                   // 0..4095
    const __bf16* src = (c2 < 2048) ? Qh : Ql;
    const unsigned db = smbase + ((c2 < 2048) ? 0u : (unsigned)(128 * 136 * 2));
    const int cc = c2 & 2047, r = cc >> 4, q = cc & 15;
    const unsigned goff = (unsigned)(((size_t)(q0 + r) * 2048 + head * HD + q * 8) * 2);
    async_b128(db + (unsigned)(r * 136 + q * 8) * 2, src, goff);
  }
  WAIT_ASYNCCNT_0;
  __syncthreads();
  v16bf qfh[4], qfl[4];
#pragma unroll
  for (int kt = 0; kt < 4; ++kt) {
    qfh[kt] = load_frag_bf16(sQh, 136, wave * 16, kt * 32, lane);
    qfl[kt] = load_frag_bf16(sQl, 136, wave * 16, kt * 32, lane);
  }

  v8f o[8] = {};
  float m_run[8], l_run[8];
#pragma unroll
  for (int v = 0; v < 8; ++v) { m_run[v] = -__builtin_inff(); l_run[v] = 0.0f; }

  __bf16* const pWh = sP + wave * (2 * 16 * 34);
  __bf16* const pWl = pWh + 16 * 34;
  const int nkt = (q0 + 128) >> 5;
  const int cn = lane & 15, chalf = lane >> 4;

  for (int ktile = 0; ktile < nkt; ++ktile) {
    const int k0 = ktile * 32;
    __syncthreads();   // prior reads done before restage
    // K tile 32x128 hi/lo (4 chunks/thread)
#pragma unroll
    for (int i = 0; i < 4; ++i) {
      const int c2 = i * 256 + tid;                 // 0..1023
      const __bf16* src = (c2 < 512) ? Kh : Kl;
      const unsigned db = smbase + ((c2 < 512) ? 0u : 8704u);
      const int cc = c2 & 511, r = cc >> 4, q = cc & 15;
      const unsigned goff = (unsigned)(((size_t)(k0 + r) * 512 + kvh * HD + q * 8) * 2);
      async_b128(db + (unsigned)(r * 136 + q * 8) * 2, src, goff);
    }
    // Vt tile 128x32 hi/lo (4 chunks/thread)
#pragma unroll
    for (int i = 0; i < 4; ++i) {
      const int c2 = i * 256 + tid;
      const __bf16* src = (c2 < 512) ? Vth : Vtl;
      const unsigned db = smbase + ((c2 < 512) ? 17408u : 27648u);
      const int cc = c2 & 511, r = cc >> 2, q = cc & 3;
      const unsigned goff = (unsigned)(((size_t)(kvh * HD + r) * T_SEQ + k0 + q * 8) * 2);
      async_b128(db + (unsigned)(r * 40 + q * 8) * 2, src, goff);
    }
    WAIT_ASYNCCNT_0;
    __syncthreads();

    // S = Q * K^T (16 q-rows x 32 keys)
    v8f accS[2] = {};
#pragma unroll
    for (int sj = 0; sj < 2; ++sj)
#pragma unroll
      for (int kt = 0; kt < 4; ++kt) {
        v16bf bh = load_frag_bf16(sKh, 136, sj * 16, kt * 32, lane);
        v16bf bl = load_frag_bf16(sKl, 136, sj * 16, kt * 32, lane);
        accS[sj] = wmma_split(qfh[kt], qfl[kt], bh, bl, accS[sj]);
      }

    // online softmax (row stats replicated across each 16-lane half)
#pragma unroll
    for (int v = 0; v < 8; ++v) {
      const int qi = qr0 + v + chalf * 8;
      float x0 = accS[0][v] * scale;
      float x1 = accS[1][v] * scale;
      if (k0 + cn      > qi) x0 = -1e30f;
      if (k0 + 16 + cn > qi) x1 = -1e30f;
      float tmax = fmaxf(x0, x1);
#pragma unroll
      for (int off = 8; off > 0; off >>= 1) tmax = fmaxf(tmax, __shfl_xor(tmax, off, 32));
      const float mn    = fmaxf(m_run[v], tmax);
      const float alpha = expf(m_run[v] - mn);
      const float p0 = expf(x0 - mn);
      const float p1 = expf(x1 - mn);
      float rs = p0 + p1;
#pragma unroll
      for (int off = 8; off > 0; off >>= 1) rs += __shfl_xor(rs, off, 32);
      l_run[v] = l_run[v] * alpha + rs;
      m_run[v] = mn;
#pragma unroll
      for (int os = 0; os < 8; ++os) o[os][v] *= alpha;
      const int pr = v + chalf * 8;
      __bf16 h, l;
      split_bf16(p0, h, l);
      pWh[pr * 34 + cn] = h;      pWl[pr * 34 + cn] = l;
      split_bf16(p1, h, l);
      pWh[pr * 34 + 16 + cn] = h; pWl[pr * 34 + 16 + cn] = l;
    }
    __syncthreads();

    // O += P * V  (16x32)*(32x128)
    v16bf pah = load_frag_bf16(pWh, 34, 0, 0, lane);
    v16bf pal = load_frag_bf16(pWl, 34, 0, 0, lane);
#pragma unroll
    for (int os = 0; os < 8; ++os) {
      v16bf vh = load_frag_bf16(sVth, 40, os * 16, 0, lane);
      v16bf vl = load_frag_bf16(sVtl, 40, os * 16, 0, lane);
      o[os] = wmma_split(pah, pal, vh, vl, o[os]);
    }
  }

  // normalize, emit bf16 hi/lo (feeds final GEMM directly)
#pragma unroll
  for (int v = 0; v < 8; ++v) {
    const float invl = 1.0f / l_run[v];
    const int row = qr0 + v + chalf * 8;
#pragma unroll
    for (int os = 0; os < 8; ++os) {
      const float y = o[os][v] * invl;
      __bf16 h, l; split_bf16(y, h, l);
      const size_t idx = (size_t)row * 2048 + head * HD + os * 16 + cn;
      Oh[idx] = h; Ol[idx] = l;
    }
  }
}

extern "C" void kernel_launch(void* const* d_in, const int* in_sizes, int n_in,
                              void* d_out, int out_size, void* d_ws, size_t ws_size,
                              hipStream_t stream) {
  const int*   positions = (const int*)d_in[0];
  const float* hidden    = (const float*)d_in[1];
  const float* w_qkv     = (const float*)d_in[2];
  const float* w_o       = (const float*)d_in[3];

  char* w = (char*)d_ws;
  float*  qkv  = (float*)w;                               w += (size_t)T_SEQ * QKV_COLS * 4;
  __bf16* Hh   = (__bf16*)w;
  __bf16* Hl   = Hh + (size_t)2048 * 2048;                w += (size_t)2048 * 2048 * 4;
  __bf16* WqTh = (__bf16*)w;
  __bf16* WqTl = WqTh + (size_t)3072 * 2048;              w += (size_t)3072 * 2048 * 4;
  __bf16* Qh   = (__bf16*)w;
  __bf16* Ql   = Qh + (size_t)2048 * 2048;                w += (size_t)2048 * 2048 * 4;
  __bf16* Kh   = (__bf16*)w;
  __bf16* Kl   = Kh + (size_t)2048 * 512;                 w += (size_t)2048 * 512 * 4;
  __bf16* Vth  = (__bf16*)w;
  __bf16* Vtl  = Vth + (size_t)512 * 2048;                w += (size_t)512 * 2048 * 4;
  __bf16* AttnH = Hh, *AttnL = Hl;     // reuse: hidden split dead after GEMM1
  __bf16* WoTh  = WqTh, *WoTl = WqTl;  // reuse: w_qkv^T dead after GEMM1

  // 1) operand prep + QKV projection
  split_kernel<<<(2048 * 2048) / 256, 256, 0, stream>>>(hidden, Hh, Hl, 2048 * 2048);
  transpose_split_kernel<<<dim3(3072 / 32, 2048 / 32), 256, 0, stream>>>(
      w_qkv, QKV_COLS, WqTh, WqTl, HIDDEN_D, HIDDEN_D, QKV_COLS);
  gemm_bf16_async_kernel<<<dim3(QKV_COLS / 128, T_SEQ / 128), 256, 0, stream>>>(
      Hh, Hl, WqTh, WqTl, qkv, T_SEQ, QKV_COLS, HIDDEN_D, QKV_COLS);
  // 2) mRoPE -> bf16 hi/lo Q,K ; V -> transposed bf16 hi/lo
  rope_split_kernel<<<T_SEQ, 256, 0, stream>>>(positions, qkv, Qh, Ql, Kh, Kl);
  for (int kvh = 0; kvh < NKV; ++kvh)
    transpose_split_kernel<<<dim3(HD / 32, T_SEQ / 32), 256, 0, stream>>>(
        qkv + 2560 + kvh * HD, QKV_COLS,
        Vth + (size_t)kvh * HD * T_SEQ, Vtl + (size_t)kvh * HD * T_SEQ,
        T_SEQ, T_SEQ, HD);
  // 3) causal GQA flash attention (emits bf16 hi/lo)
  attn_kernel<<<dim3(T_SEQ / 128, NH), 256, 0, stream>>>(
      Qh, Ql, Kh, Kl, Vth, Vtl, AttnH, AttnL);
  // 4) output projection
  transpose_split_kernel<<<dim3(2048 / 32, 2048 / 32), 256, 0, stream>>>(
      w_o, HIDDEN_D, WoTh, WoTl, HIDDEN_D, HIDDEN_D, HIDDEN_D);
  gemm_bf16_async_kernel<<<dim3(HIDDEN_D / 128, T_SEQ / 128), 256, 0, stream>>>(
      AttnH, AttnL, WoTh, WoTl, (float*)d_out, T_SEQ, HIDDEN_D, NH * HD, HIDDEN_D);
}
